// UCM_NetV2BNN_18906446037728
// MI455X (gfx1250) — compile-verified, matrix operand order checked by reading, not run on hardware
//
#include <hip/hip_runtime.h>

typedef __attribute__((ext_vector_type(16))) _Float16 v16h;
typedef __attribute__((ext_vector_type(8)))  float    v8f;

#define EPSF 1e-5f
#define MAXC 64

static __device__ __forceinline__ int iclampi(int v, int lo, int hi) {
  return v < lo ? lo : (v > hi ? hi : v);
}

// ---------------------------------------------------------------------------
// LayerNorm over a per-thread channel vector
// ---------------------------------------------------------------------------
static __device__ __forceinline__ void ln_apply(float* v, int C,
                                                const float* w, const float* b) {
  float m = 0.f;
  for (int c = 0; c < C; ++c) m += v[c];
  m /= (float)C;
  float var = 0.f;
  for (int c = 0; c < C; ++c) { float d = v[c] - m; var += d * d; }
  var /= (float)C;
  float rs = rsqrtf(var + EPSF);
  for (int c = 0; c < C; ++c) v[c] = (v[c] - m) * rs * w[c] + b[c];
}

// ---------------------------------------------------------------------------
// XNOR FC weight prep, pre-swizzled into WMMA B-operand register layout.
// Logical B[k][n] = sign(W[n,k]) * alpha[n], zero-padded to 64x64.
// Output index: (((ct*2 + q)*32 + lane)*16 + i); for lane (khi = lane&16 ? 8:0,
// col = ct*16 + (lane&15)):  i<8 -> k = 32q+khi+i ; i>=8 -> k = 32q+khi+8+i.
// The FC kernel loads each v16h B operand as one aligned 32B vector load.
// ---------------------------------------------------------------------------
__global__ void prep_fc(const float* __restrict__ W, int C, _Float16* __restrict__ WB) {
  __shared__ float alpha[MAXC];
  int t = threadIdx.x;
  if (t < C) {
    float a = 0.f;
    for (int k = 0; k < C; ++k) a += fabsf(W[t * C + k]);
    alpha[t] = a / (float)C;
  }
  __syncthreads();
  for (int idx = t; idx < 64 * 64; idx += blockDim.x) {
    int i    = idx & 15;
    int lane = (idx >> 4) & 31;
    int q    = (idx >> 9) & 1;
    int ct   = idx >> 10;
    int khi  = (lane & 16) ? 8 : 0;
    int col  = ct * 16 + (lane & 15);
    int k    = 32 * q + khi + ((i < 8) ? i : (8 + i));  // i>=8: 16 + (i-8)
    float v = 0.f;
    if (k < C && col < C) {
      float w = W[col * C + k];
      v = (w > 0.f) ? alpha[col] : ((w < 0.f) ? -alpha[col] : 0.f);
    }
    WB[idx] = (_Float16)v;
  }
}

// Generic XNOR conv weight prep: Wp[o*inner+i] = sign(W[o,i]) * mean|W[o,:]|
__global__ void prep_conv(const float* __restrict__ W, int Co, int inner,
                          float* __restrict__ Wp) {
  __shared__ float alpha[MAXC];
  int t = threadIdx.x;
  if (t < Co) {
    float a = 0.f;
    for (int i = 0; i < inner; ++i) a += fabsf(W[t * inner + i]);
    alpha[t] = a / (float)inner;
  }
  __syncthreads();
  for (int idx = t; idx < Co * inner; idx += blockDim.x) {
    int o = idx / inner;
    float w = W[idx];
    Wp[idx] = (w > 0.f) ? alpha[o] : ((w < 0.f) ? -alpha[o] : 0.f);
  }
}

// ---------------------------------------------------------------------------
// enc1: 3x3 conv (raw weights) value, computed on the fly
// ---------------------------------------------------------------------------
static __device__ __forceinline__ float enc1_val(const float* __restrict__ x,
                                                 const float* __restrict__ w,
                                                 const float* __restrict__ bias,
                                                 int b, int c, int h, int wx) {
  const int H = 256, W = 256;
  float s = bias[c];
  for (int ci = 0; ci < 3; ++ci) {
    const float* xp = x + ((long)(b * 3 + ci)) * H * W;
    const float* wp = w + (c * 3 + ci) * 9;
    for (int ky = 0; ky < 3; ++ky) {
      int y = h + ky - 1;
      if (y < 0 || y >= H) continue;
      for (int kx = 0; kx < 3; ++kx) {
        int xx = wx + kx - 1;
        if (xx < 0 || xx >= W) continue;
        s += wp[ky * 3 + kx] * xp[y * W + xx];
      }
    }
  }
  return s;
}

__global__ void enc1_gn_stats(const float* __restrict__ x, const float* __restrict__ w,
                              const float* __restrict__ bias,
                              float* __restrict__ mean, float* __restrict__ rstd) {
  const int H = 256, W = 256;
  int b = blockIdx.x >> 2, g = blockIdx.x & 3;
  __shared__ float sh_s[256], sh_q[256];
  float s = 0.f, q = 0.f;
  for (int i = threadIdx.x; i < 2 * H * W; i += blockDim.x) {
    int c = 2 * g + (i >= H * W ? 1 : 0);
    int pix = i & (H * W - 1);
    float v = enc1_val(x, w, bias, b, c, pix / W, pix % W);
    s += v; q += v * v;
  }
  sh_s[threadIdx.x] = s; sh_q[threadIdx.x] = q;
  __syncthreads();
  for (int st = 128; st > 0; st >>= 1) {
    if ((int)threadIdx.x < st) {
      sh_s[threadIdx.x] += sh_s[threadIdx.x + st];
      sh_q[threadIdx.x] += sh_q[threadIdx.x + st];
    }
    __syncthreads();
  }
  if (threadIdx.x == 0) {
    float n = (float)(2 * H * W);
    float m = sh_s[0] / n;
    float var = sh_q[0] / n - m * m;
    mean[blockIdx.x] = m;
    rstd[blockIdx.x] = rsqrtf(var + EPSF);
  }
}

__global__ void enc1_gn_pool_relu(const float* __restrict__ x, const float* __restrict__ w,
                                  const float* __restrict__ bias,
                                  const float* __restrict__ mean, const float* __restrict__ rstd,
                                  const float* __restrict__ gw, const float* __restrict__ gb,
                                  float* __restrict__ out) {
  const int Ho = 128, Wo = 128;
  long idx = (long)blockIdx.x * blockDim.x + threadIdx.x;
  if (idx >= (long)32 * 8 * Ho * Wo) return;
  int wo = idx % Wo; long t = idx / Wo;
  int ho = t % Ho; t /= Ho;
  int c = t % 8; int b = (int)(t / 8);
  int g = c >> 1;
  float m = mean[b * 4 + g], rs = rstd[b * 4 + g];
  float ga = gw[c], be = gb[c];
  float best = -1e30f;
  for (int dy = 0; dy < 2; ++dy)
    for (int dx = 0; dx < 2; ++dx) {
      float v = enc1_val(x, w, bias, b, c, 2 * ho + dy, 2 * wo + dx);
      v = (v - m) * rs * ga + be;
      best = fmaxf(best, v);
    }
  out[idx] = fmaxf(best, 0.f);
}

// ---------------------------------------------------------------------------
// Patch-embed (1x1 stride-2 XNOR conv) + pe-LN + block-LN -> x1 tokens (B,N,C)
// ---------------------------------------------------------------------------
__global__ void pe_tokens_x1(const float* __restrict__ in, int Ci, int Hi, int Wi,
                             const float* __restrict__ Wp, const float* __restrict__ bias,
                             const float* __restrict__ lnw, const float* __restrict__ lnb,
                             const float* __restrict__ nw, const float* __restrict__ nb,
                             float* __restrict__ x1, int C, int B) {
  int Ho = Hi >> 1, Wo = Wi >> 1, N = Ho * Wo;
  long t = (long)blockIdx.x * blockDim.x + threadIdx.x;
  if (t >= (long)B * N) return;
  int n = (int)(t % N); int b = (int)(t / N);
  int h = n / Wo, w_ = n % Wo;
  const float* ip = in + ((long)b * Ci) * Hi * Wi + (long)(2 * h) * Wi + 2 * w_;
  float xin[MAXC];
  for (int ci = 0; ci < Ci; ++ci) xin[ci] = ip[(long)ci * Hi * Wi];
  float v[MAXC];
  for (int co = 0; co < C; ++co) {
    float s = bias[co];
    const float* wp = Wp + co * Ci;
    for (int ci = 0; ci < Ci; ++ci) s += wp[ci] * xin[ci];
    v[co] = s;
  }
  ln_apply(v, C, lnw, lnb);
  ln_apply(v, C, nw, nb);
  float* op = x1 + t * (long)C;
  for (int c = 0; c < C; ++c) op[c] = v[c];
}

// Decoder path: map (B,C,N) -> block-LN -> x1 tokens
__global__ void map_tokens_x1(const float* __restrict__ in,
                              const float* __restrict__ nw, const float* __restrict__ nb,
                              float* __restrict__ x1, int C, int N, int B) {
  long t = (long)blockIdx.x * blockDim.x + threadIdx.x;
  if (t >= (long)B * N) return;
  int n = (int)(t % N); int b = (int)(t / N);
  float v[MAXC];
  for (int c = 0; c < C; ++c) v[c] = in[((long)b * C + c) * N + n];
  ln_apply(v, C, nw, nb);
  float* op = x1 + t * (long)C;
  for (int c = 0; c < C; ++c) op[c] = v[c];
}

// tokens (B,N,C) -> LN -> NCHW map
__global__ void token_ln_map(const float* __restrict__ tkn,
                             const float* __restrict__ nw, const float* __restrict__ nb,
                             float* __restrict__ map, int C, int N, int B) {
  long t = (long)blockIdx.x * blockDim.x + threadIdx.x;
  if (t >= (long)B * N) return;
  int n = (int)(t % N); int b = (int)(t / N);
  float v[MAXC];
  const float* ip = tkn + t * (long)C;
  for (int c = 0; c < C; ++c) v[c] = ip[c];
  ln_apply(v, C, nw, nb);
  for (int c = 0; c < C; ++c) map[((long)b * C + c) * N + n] = v[c];
}

// ---------------------------------------------------------------------------
// UCM dual-FC WMMA kernel: xa/x2 = x1 @ sign(W)^T*alpha + b + x1,
// written into concat map xc (B, 2C, N). 4 waves/block, 16 tokens per wave.
// Structure exploited:
//  - BN is always a multiple of 64 -> no row guards/clamps needed
//  - N is a power of two (nshift = log2 N) -> bimg/n via shift/mask
//  - the 8 rows per accumulator never cross an image boundary -> per lane the
//    8 outputs are contiguous & 32B aligned -> two float4 stores per tile.
// ---------------------------------------------------------------------------
__global__ void __launch_bounds__(128)
ucm_fc_wmma(const float* __restrict__ x1,
            const _Float16* __restrict__ WB1, const _Float16* __restrict__ WB2,
            const float* __restrict__ bias1, const float* __restrict__ bias2,
            float* __restrict__ xc, int C, int Nimg, int nshift) {
  const int lane = threadIdx.x & 31;
  const int wave = threadIdx.x >> 5;
  const long rowBase = (long)blockIdx.x * 64 + (long)wave * 16;

  const int m   = lane & 15;
  const int khi = (lane & 16) ? 8 : 0;
  const long r = rowBase + m;
  const float* xr = x1 + r * (long)C;  // 32B-aligned: C % 8 == 0

  const int kcnt = (C > 32) ? 2 : 1;
  v16h A[2];
#pragma unroll
  for (int q = 0; q < 2; ++q) {
    float4 r0 = make_float4(0.f, 0.f, 0.f, 0.f), r1 = r0, r2 = r0, r3 = r0;
    int s0 = 32 * q + khi;       // run of 8 floats, fully valid iff s0 < C
    if (s0 < C) {
      const float4* p = (const float4*)(xr + s0);
      r0 = p[0]; r1 = p[1];
    }
    int s1 = s0 + 16;
    if (s1 < C) {
      const float4* p = (const float4*)(xr + s1);
      r2 = p[0]; r3 = p[1];
    }
    A[q][0]  = (_Float16)r0.x; A[q][1]  = (_Float16)r0.y;
    A[q][2]  = (_Float16)r0.z; A[q][3]  = (_Float16)r0.w;
    A[q][4]  = (_Float16)r1.x; A[q][5]  = (_Float16)r1.y;
    A[q][6]  = (_Float16)r1.z; A[q][7]  = (_Float16)r1.w;
    A[q][8]  = (_Float16)r2.x; A[q][9]  = (_Float16)r2.y;
    A[q][10] = (_Float16)r2.z; A[q][11] = (_Float16)r2.w;
    A[q][12] = (_Float16)r3.x; A[q][13] = (_Float16)r3.y;
    A[q][14] = (_Float16)r3.z; A[q][15] = (_Float16)r3.w;
  }

  const int mh = (lane & 16) ? 8 : 0;
  const long rEp   = rowBase + mh;          // 8-aligned run inside one image
  const long bimg  = rEp >> nshift;
  const long n0    = rEp & (long)(Nimg - 1);
  const float* resp = x1 + rEp * (long)C;   // residual rows base

  const int nct = (C + 15) >> 4;
#pragma unroll
  for (int f = 0; f < 2; ++f) {
    const v16h* WBv = (const v16h*)(f ? WB2 : WB1);  // swizzled: [(ct*2+q)*32 + lane]
    const float* bs = f ? bias2 : bias1;
    const int chOff = f ? C : 0;
    for (int ct = 0; ct < nct; ++ct) {
      const int col = ct * 16 + (lane & 15);
      v16h Bv = WBv[(ct * 2 + 0) * 32 + lane];
      v8f acc = {};
      acc = __builtin_amdgcn_wmma_f32_16x16x32_f16(false, A[0], false, Bv,
                                                   (short)0, acc, false, false);
      if (kcnt == 2) {  // uniform
        v16h Bv2 = WBv[(ct * 2 + 1) * 32 + lane];
        acc = __builtin_amdgcn_wmma_f32_16x16x32_f16(false, A[1], false, Bv2,
                                                     (short)0, acc, false, false);
      }
      if (col < C) {
        const float bv = bs[col];
        float* dst = xc + (bimg * 2 * C + chOff + col) * (long)Nimg + n0;
        float4 o0, o1;
        o0.x = acc[0] + bv + resp[0 * C + col];
        o0.y = acc[1] + bv + resp[1 * C + col];
        o0.z = acc[2] + bv + resp[2 * C + col];
        o0.w = acc[3] + bv + resp[3 * C + col];
        o1.x = acc[4] + bv + resp[4 * C + col];
        o1.y = acc[5] + bv + resp[5 * C + col];
        o1.z = acc[6] + bv + resp[6 * C + col];
        o1.w = acc[7] + bv + resp[7 * C + col];
        ((float4*)dst)[0] = o0;
        ((float4*)dst)[1] = o1;
      }
    }
  }
}

// ---------------------------------------------------------------------------
// Generic region mean/rstd reduction (ln_hw planes and GroupNorm groups)
// ---------------------------------------------------------------------------
__global__ void region_stats(const float* __restrict__ src, long rlen,
                             float* __restrict__ mean, float* __restrict__ rstd) {
  __shared__ float sh_s[256], sh_q[256];
  long base = (long)blockIdx.x * rlen;
  float s = 0.f, q = 0.f;
  for (long i = threadIdx.x; i < rlen; i += blockDim.x) {
    float v = src[base + i]; s += v; q += v * v;
  }
  sh_s[threadIdx.x] = s; sh_q[threadIdx.x] = q;
  __syncthreads();
  for (int st = 128; st > 0; st >>= 1) {
    if ((int)threadIdx.x < st) {
      sh_s[threadIdx.x] += sh_s[threadIdx.x + st];
      sh_q[threadIdx.x] += sh_q[threadIdx.x + st];
    }
    __syncthreads();
  }
  if (threadIdx.x == 0) {
    float m = sh_s[0] / (float)rlen;
    float var = sh_q[0] / (float)rlen - m * m;
    mean[blockIdx.x] = m;
    rstd[blockIdx.x] = rsqrtf(var + EPSF);
  }
}

// ---------------------------------------------------------------------------
// ln_hw + grouped 3x3 XNOR dwconv + residual + x2 doubling -> t1 tokens
// ---------------------------------------------------------------------------
__global__ void dwconv_t1(const float* __restrict__ xc, const float* __restrict__ mean,
                          const float* __restrict__ rstd, const float* __restrict__ Wd,
                          const float* __restrict__ bd, const float* __restrict__ x1,
                          float* __restrict__ t1, int C, int H, int W, int B) {
  int N = H * W;
  long idx = (long)blockIdx.x * blockDim.x + threadIdx.x;
  if (idx >= (long)B * C * N) return;
  int n = (int)(idx % N); long t = idx / N;
  int c = (int)(t % C); int b = (int)(t / C);
  int h = n / W, w_ = n % W;
  float s = bd[c];
#pragma unroll
  for (int ic = 0; ic < 2; ++ic) {
    int ch = 2 * c + ic;
    long pl = (long)b * 2 * C + ch;
    float m = mean[pl], rs = rstd[pl];
    const float* sp = xc + pl * (long)N;
    const float* wp = Wd + c * 18 + ic * 9;
    for (int ky = 0; ky < 3; ++ky) {
      int y = h + ky - 1;
      if (y < 0 || y >= H) continue;
      for (int kx = 0; kx < 3; ++kx) {
        int x = w_ + kx - 1;
        if (x < 0 || x >= W) continue;
        s += wp[ky * 3 + kx] * ((sp[y * W + x] - m) * rs);
      }
    }
  }
  long ti = ((long)b * N + n) * C + c;
  t1[ti] = 2.f * (s + x1[ti]);
}

// ---------------------------------------------------------------------------
// Decoder 1x1 XNOR conv over NCHW map
// ---------------------------------------------------------------------------
__global__ void conv1x1_xnor(const float* __restrict__ in, const float* __restrict__ Wp,
                             const float* __restrict__ bias, float* __restrict__ out,
                             int B, int Ci, int Co, int N) {
  long idx = (long)blockIdx.x * blockDim.x + threadIdx.x;
  if (idx >= (long)B * Co * N) return;
  int n = (int)(idx % N); long t = idx / N;
  int co = (int)(t % Co); int b = (int)(t / Co);
  const float* ip = in + ((long)b * Ci) * N + n;
  const float* wp = Wp + co * Ci;
  float s = bias[co];
  for (int ci = 0; ci < Ci; ++ci) s += wp[ci] * ip[(long)ci * N];
  out[((long)b * Co + co) * N + n] = s;
}

// ---------------------------------------------------------------------------
// Bilinear 2x upsample (half-pixel centers, clamped)
// ---------------------------------------------------------------------------
static __device__ __forceinline__ float bilerp2x(const float* __restrict__ p,
                                                 int H, int W, int oy, int ox) {
  float fy = oy * 0.5f - 0.25f, fx = ox * 0.5f - 0.25f;
  int y0 = (int)floorf(fy), x0 = (int)floorf(fx);
  float wy = fy - (float)y0, wx = fx - (float)x0;
  int y0c = iclampi(y0, 0, H - 1), y1c = iclampi(y0 + 1, 0, H - 1);
  int x0c = iclampi(x0, 0, W - 1), x1c = iclampi(x0 + 1, 0, W - 1);
  float v00 = p[y0c * W + x0c], v01 = p[y0c * W + x1c];
  float v10 = p[y1c * W + x0c], v11 = p[y1c * W + x1c];
  return (1.f - wy) * ((1.f - wx) * v00 + wx * v01) +
         wy * ((1.f - wx) * v10 + wx * v11);
}

// GN (affine) + up2 + ReLU + skip-add (affine commutes with bilerp)
__global__ void gn_up_relu_add(const float* __restrict__ xin,
                               const float* __restrict__ mean, const float* __restrict__ rstd,
                               const float* __restrict__ gw, const float* __restrict__ gb,
                               const float* __restrict__ skip, float* __restrict__ out,
                               int B, int C, int h, int w) {
  int Ho = 2 * h, Wo = 2 * w;
  long idx = (long)blockIdx.x * blockDim.x + threadIdx.x;
  if (idx >= (long)B * C * Ho * Wo) return;
  int ox = (int)(idx % Wo); long t = idx / Wo;
  int oy = (int)(t % Ho); t /= Ho;
  int c = (int)(t % C); int b = (int)(t / C);
  const float* p = xin + ((long)b * C + c) * (long)(h * w);
  float raw = bilerp2x(p, h, w, oy, ox);
  int g = c / (C / 4);
  float m = mean[b * 4 + g], rs = rstd[b * 4 + g];
  float v = (raw - m) * rs * gw[c] + gb[c];
  out[idx] = fmaxf(v, 0.f) + skip[idx];
}

// dec5 tail: up2 + ReLU (per channel) + final 1x1 conv -> d_out
__global__ void up_relu_final(const float* __restrict__ xin,
                              const float* __restrict__ fw, const float* __restrict__ fb,
                              float* __restrict__ out, int B) {
  const int h = 128, w = 128, Ho = 256, Wo = 256;
  long idx = (long)blockIdx.x * blockDim.x + threadIdx.x;
  if (idx >= (long)B * Ho * Wo) return;
  int ox = (int)(idx % Wo); long t = idx / Wo;
  int oy = (int)(t % Ho); int b = (int)(t / Ho);
  float s = fb[0];
  for (int c = 0; c < 3; ++c) {
    const float* p = xin + ((long)b * 3 + c) * (long)(h * w);
    s += fw[c] * fmaxf(bilerp2x(p, h, w, oy, ox), 0.f);
  }
  out[idx] = s;
}

// ---------------------------------------------------------------------------
// Host orchestration
// ---------------------------------------------------------------------------
static inline unsigned cdiv(long n, int d) { return (unsigned)((n + d - 1) / d); }
static inline int ilog2i(int v) { int s = 0; while ((1 << s) < v) ++s; return s; }

extern "C" void kernel_launch(void* const* d_in, const int* in_sizes, int n_in,
                              void* d_out, int out_size, void* d_ws, size_t ws_size,
                              hipStream_t stream) {
  (void)in_sizes; (void)n_in; (void)out_size; (void)ws_size;
  const int B = 32;
  const float* X = (const float*)d_in[0];
  auto F = [&](int i) { return (const float*)d_in[i]; };

  // Workspace carving (floats, 256B-aligned chunks)
  float* W = (float*)d_ws;
  size_t o = 0;
  auto A = [&](size_t n) { float* p = W + o; o += (n + 63) & ~((size_t)63); return p; };
  float* S0 = A((size_t)B * 8 * 128 * 128);
  float* S1 = A((size_t)B * 16 * 64 * 64);
  float* S2 = A((size_t)B * 24 * 32 * 32);
  float* S3 = A((size_t)B * 32 * 16 * 16);
  float* S4 = A((size_t)B * 48 * 8 * 8);
  float* P2 = A(4300800);   // x1 tokens        (max 4,194,304)
  float* P3 = A(8601600);   // xc concat map    (max 8,388,608)
  float* P4 = A(4300800);   // t1 tokens
  float* P5 = A(4300800);   // current out map
  float* PC = A(1600000);   // decoder conv out (max 1,572,864)
  float* MEAN = A(8192);
  float* RSTD = A(8192);
  _Float16* WB1 = (_Float16*)A(2112);  // 4096 f16, swizzled B layout
  _Float16* WB2 = (_Float16*)A(2112);
  float* WPC = A(4096);     // prepared conv weights
  float* WPD = A(2048);     // prepared dw weights

  // --- enc1: 3x3 conv + GN(4) + maxpool2 + ReLU -> S0 (32,8,128,128)
  enc1_gn_stats<<<B * 4, 256, 0, stream>>>(X, F(1), F(2), MEAN, RSTD);
  enc1_gn_pool_relu<<<cdiv((long)B * 8 * 128 * 128, 256), 256, 0, stream>>>(
      X, F(1), F(2), MEAN, RSTD, F(3), F(4), S0);

  // --- encoder stages
  const int eCi[5] = {8, 16, 24, 32, 48};
  const int eC[5]  = {16, 24, 32, 48, 64};
  const int eHi[5] = {128, 64, 32, 16, 8};
  float* eIn[5]  = {S0, S1, S2, S3, S4};
  float* eOut[5] = {S1, S2, S3, S4, P5};
  for (int i = 0; i < 5; ++i) {
    int Ci = eCi[i], C = eC[i], Hi = eHi[i], Ho = Hi / 2, N = Ho * Ho;
    long BN = (long)B * N;
    int peW = 5 + 4 * i, bb = 25 + 8 * i, nrm = 105 + 2 * i;
    prep_conv<<<1, 256, 0, stream>>>(F(peW), C, Ci, WPC);
    prep_fc<<<1, 256, 0, stream>>>(F(bb + 2), C, WB1);
    prep_fc<<<1, 256, 0, stream>>>(F(bb + 4), C, WB2);
    pe_tokens_x1<<<cdiv(BN, 256), 256, 0, stream>>>(
        eIn[i], Ci, Hi, Hi, WPC, F(peW + 1), F(peW + 2), F(peW + 3),
        F(bb + 0), F(bb + 1), P2, C, B);
    ucm_fc_wmma<<<cdiv(BN, 64), 128, 0, stream>>>(
        P2, WB1, WB2, F(bb + 3), F(bb + 5), P3, C, N, ilog2i(N));
    region_stats<<<B * 2 * C, 256, 0, stream>>>(P3, (long)N, MEAN, RSTD);
    prep_conv<<<1, 256, 0, stream>>>(F(bb + 6), C, 18, WPD);
    dwconv_t1<<<cdiv((long)B * C * N, 256), 256, 0, stream>>>(
        P3, MEAN, RSTD, WPD, F(bb + 7), P2, P4, C, Ho, Ho, B);
    token_ln_map<<<cdiv(BN, 256), 256, 0, stream>>>(
        P4, F(nrm), F(nrm + 1), eOut[i], C, N, B);
  }

  // --- decoder stages
  const int dCi[5] = {64, 48, 32, 24, 16};
  const int dCo[5] = {48, 32, 24, 16, 8};
  const int dh[5]  = {4, 8, 16, 32, 64};
  float* dSkip[5] = {S4, S3, S2, S1, S0};
  for (int i = 0; i < 5; ++i) {
    int Ci = dCi[i], C = dCo[i], h = dh[i], Hs = 2 * h, N = Hs * Hs;
    long BN = (long)B * N;
    int decW = 125 + 2 * i, dbn = 137 + 2 * i, bb = 65 + 8 * i, nrm = 115 + 2 * i;
    prep_conv<<<1, 256, 0, stream>>>(F(decW), C, Ci, WPC);
    conv1x1_xnor<<<cdiv((long)B * C * h * h, 256), 256, 0, stream>>>(
        P5, WPC, F(decW + 1), PC, B, Ci, C, h * h);
    region_stats<<<B * 4, 256, 0, stream>>>(PC, (long)(C / 4) * h * h, MEAN, RSTD);
    gn_up_relu_add<<<cdiv((long)B * C * N, 256), 256, 0, stream>>>(
        PC, MEAN, RSTD, F(dbn), F(dbn + 1), dSkip[i], P5, B, C, h, h);
    prep_fc<<<1, 256, 0, stream>>>(F(bb + 2), C, WB1);
    prep_fc<<<1, 256, 0, stream>>>(F(bb + 4), C, WB2);
    map_tokens_x1<<<cdiv(BN, 256), 256, 0, stream>>>(
        P5, F(bb + 0), F(bb + 1), P2, C, N, B);
    ucm_fc_wmma<<<cdiv(BN, 64), 128, 0, stream>>>(
        P2, WB1, WB2, F(bb + 3), F(bb + 5), P3, C, N, ilog2i(N));
    region_stats<<<B * 2 * C, 256, 0, stream>>>(P3, (long)N, MEAN, RSTD);
    prep_conv<<<1, 256, 0, stream>>>(F(bb + 6), C, 18, WPD);
    dwconv_t1<<<cdiv((long)B * C * N, 256), 256, 0, stream>>>(
        P3, MEAN, RSTD, WPD, F(bb + 7), P2, P4, C, Hs, Hs, B);
    token_ln_map<<<cdiv(BN, 256), 256, 0, stream>>>(
        P4, F(nrm), F(nrm + 1), P5, C, N, B);
  }

  // --- dec5 (1x1 XNOR 8->3) + up2 + ReLU + final 1x1 conv -> d_out
  prep_conv<<<1, 256, 0, stream>>>(F(135), 3, 8, WPC);
  conv1x1_xnor<<<cdiv((long)B * 3 * 128 * 128, 256), 256, 0, stream>>>(
      P5, WPC, F(136), PC, B, 8, 3, 128 * 128);
  up_relu_final<<<cdiv((long)B * 256 * 256, 256), 256, 0, stream>>>(
      PC, F(147), F(148), (float*)d_out, B);
}
